// TransformNet_29033978921715
// MI455X (gfx1250) — matrix operand order repaired and checked by reading, not compiled, not run on hardware
//
#include <hip/hip_runtime.h>

typedef __attribute__((ext_vector_type(16))) _Float16 v16h;
typedef __attribute__((ext_vector_type(8)))  float    v8f;

#define KNN 20
#define EPSV 1e-5f
#define NEG 0.2f

constexpr int B_ = 8, C_ = 6, N_ = 4096;
constexpr int C1 = 64, C2 = 128, C3 = 1024;
constexpr int M1 = B_ * N_ * KNN;   // 655360 rows (conv1/conv2)
constexpr int M3 = B_ * N_;         // 32768 rows (conv3)

__device__ __forceinline__ float lrelu(float v) { return v >= 0.f ? v : NEG * v; }
// wave32 WMMA 16-bit A-matrix 16x32 per-lane K index for v16h element i, lane-group g
__device__ __forceinline__ int amap(int i, int g) { return (i < 8) ? (g * 8 + i) : (16 + g * 8 + (i - 8)); }

// CDNA5 async global->LDS copy (ASYNCcnt-tracked, bypasses VGPRs).
// LDS flat pointers carry the LDS byte offset in bits [31:0].
__device__ __forceinline__ void async_b128(uint32_t ldsoff, uint64_t gaddr) {
  asm volatile("global_load_async_to_lds_b128 %0, %1, off" :: "v"(ldsoff), "v"(gaddr) : "memory");
}
__device__ __forceinline__ void wait_async0() {
  asm volatile("s_wait_asynccnt 0x0" ::: "memory");
}

// ---------------- kNN (top-20 by pd = 2*inner - |xj|^2, row-constant dropped) ----------------
__global__ void knn_kernel(const float* __restrict__ x, int* __restrict__ idx) {
  __shared__ float tile[256][C_];
  int b = blockIdx.y;
  int q = blockIdx.x * 256 + threadIdx.x;
  float p[C_];
#pragma unroll
  for (int c = 0; c < C_; ++c) p[c] = x[(b * C_ + c) * N_ + q];
  float bestv[KNN]; int besti[KNN];
#pragma unroll
  for (int s = 0; s < KNN; ++s) { bestv[s] = -3.4e38f; besti[s] = 0; }
  for (int t0 = 0; t0 < N_; t0 += 256) {
    __syncthreads();
#pragma unroll
    for (int c = 0; c < C_; ++c) tile[threadIdx.x][c] = x[(b * C_ + c) * N_ + t0 + threadIdx.x];
    __syncthreads();
    for (int j = 0; j < 256; ++j) {
      float inner = 0.f, qq = 0.f;
#pragma unroll
      for (int c = 0; c < C_; ++c) { float w = tile[j][c]; inner += p[c] * w; qq += w * w; }
      float v = 2.f * inner - qq;
      if (v > bestv[KNN - 1]) {
        float cv = v; int ci = t0 + j;
#pragma unroll
        for (int s = 0; s < KNN; ++s) {
          if (cv > bestv[s]) { float tv = bestv[s]; int ti = besti[s]; bestv[s] = cv; besti[s] = ci; cv = tv; ci = ti; }
        }
      }
    }
  }
#pragma unroll
  for (int s = 0; s < KNN; ++s) idx[(b * N_ + q) * KNN + s] = besti[s];
}

// ---------------- conv1: gather edge features + WMMA (K=12 padded to 32) + stats ----------------
__global__ void conv1_kernel(const float* __restrict__ x, const int* __restrict__ idx,
                             const float* __restrict__ w1, _Float16* __restrict__ h1,
                             float* __restrict__ stats) {
  __shared__ __align__(16) _Float16 Ws[C1][32];
  __shared__ __align__(16) _Float16 As[8][16][32];
  __shared__ __align__(16) _Float16 Cs[8][16][C1];
  __shared__ float st[2][C1];
  int tid = threadIdx.x;
  for (int i = tid; i < C1 * 32; i += 256) {
    int o = i >> 5, k = i & 31;
    Ws[o][k] = (k < 12) ? (_Float16)w1[o * 12 + k] : (_Float16)0.f;
  }
  if (tid < 2 * C1) st[tid >> 6][tid & 63] = 0.f;
  __syncthreads();
  int wv = tid >> 5, l = tid & 31, g = l >> 4, ln = l & 15;
  int m0 = (blockIdx.x * 8 + wv) * 16;
  if (l < 16) {
    int m = m0 + l;
    int b = m / (N_ * KNN); int r = m % (N_ * KNN);
    int n = r / KNN; int k = r % KNN;
    int j = idx[(b * N_ + n) * KNN + k];
#pragma unroll
    for (int c = 0; c < C_; ++c) {
      float ctr = x[(b * C_ + c) * N_ + n];
      float nb  = x[(b * C_ + c) * N_ + j];
      As[wv][l][c]      = (_Float16)(nb - ctr);
      As[wv][l][C_ + c] = (_Float16)ctr;
    }
#pragma unroll
    for (int c = 12; c < 32; ++c) As[wv][l][c] = (_Float16)0.f;
  }
  __syncthreads();
  v16h a;
#pragma unroll
  for (int i = 0; i < 16; ++i) a[i] = As[wv][ln][g * 16 + i];
  for (int nt = 0; nt < 4; ++nt) {
    v16h bm;
#pragma unroll
    for (int i = 0; i < 16; ++i) bm[i] = Ws[nt * 16 + ln][g * 16 + i];
    v8f c = {};
    c = __builtin_amdgcn_wmma_f32_16x16x32_f16(false, a, false, bm, (short)0, c, false, false);
    int oc = nt * 16 + ln;
    float s = 0.f, q = 0.f;
#pragma unroll
    for (int r = 0; r < 8; ++r) {
      float v = c[r];
      s += v; q += v * v;
      Cs[wv][g * 8 + r][oc] = (_Float16)v;
    }
    atomicAdd(&st[0][oc], s);
    atomicAdd(&st[1][oc], q);
  }
  __syncthreads();
  // coalesced b128 store of all 8 tiles (16 rows x 64 halves each)
  for (int cix = tid; cix < 1024; cix += 256) {
    int w2 = cix >> 7;
    int r  = (cix >> 3) & 15;
    int co = (cix & 7) * 8;
    int m  = (blockIdx.x * 8 + w2) * 16 + r;
    *(uint4*)&h1[(size_t)m * C1 + co] = *(const uint4*)&Cs[w2][r][co];
  }
  if (tid < C1) { atomicAdd(&stats[tid], st[0][tid]); atomicAdd(&stats[C1 + tid], st[1][tid]); }
}

// ---------------- f16 GEMM: async double-buffered A, async W, coalesced C store, stats ----------------
template<int KD, int TPB>
__global__ void gemm_async(const _Float16* __restrict__ Ain, const _Float16* __restrict__ Wh,
                           _Float16* __restrict__ Y, float* __restrict__ stats, int Nout) {
  __shared__ __align__(16) _Float16 Ws[128][KD];
  __shared__ __align__(16) _Float16 As[2][16][KD];
  __shared__ __align__(16) _Float16 Cs[16][128];
  __shared__ float st[2][128];
  int tid = threadIdx.x;
  int obase = blockIdx.y * 128;
  st[tid >> 7][tid & 127] = 0.f;
  // async-stage W slice (contiguous 128*KD halves)
  {
    constexpr int chunks = 128 * KD * 2 / 16;
    uint64_t wsrc = (uint64_t)(uintptr_t)(Wh + (size_t)obase * KD);
    uint32_t wdst = (uint32_t)(uintptr_t)&Ws[0][0];
    for (int c = tid; c < chunks; c += 256) async_b128(wdst + c * 16, wsrc + c * 16);
  }
  constexpr int achunks = 16 * KD * 2 / 16;
  int mtile0 = blockIdx.x * TPB;
  // async-stage A tile 0 (contiguous 16*KD halves)
  {
    uint64_t asrc = (uint64_t)(uintptr_t)(Ain + (size_t)mtile0 * 16 * KD);
    uint32_t adst = (uint32_t)(uintptr_t)&As[0][0][0];
    for (int c = tid; c < achunks; c += 256) async_b128(adst + c * 16, asrc + c * 16);
  }
  wait_async0();
  __syncthreads();
  int wv = tid >> 5, l = tid & 31, g = l >> 4, ln = l & 15;
  for (int it = 0; it < TPB; ++it) {
    int cur = it & 1;
    if (it + 1 < TPB) {  // prefetch next A tile while computing this one
      uint64_t asrc = (uint64_t)(uintptr_t)(Ain + (size_t)(mtile0 + it + 1) * 16 * KD);
      uint32_t adst = (uint32_t)(uintptr_t)&As[cur ^ 1][0][0];
      for (int c = tid; c < achunks; c += 256) async_b128(adst + c * 16, asrc + c * 16);
    }
    int cb = wv * 16;
    v8f c8 = {};
#pragma unroll
    for (int ks = 0; ks < KD / 32; ++ks) {
      v16h a, bm;
#pragma unroll
      for (int i = 0; i < 16; ++i) a[i] = As[cur][ln][ks * 32 + g * 16 + i];
#pragma unroll
      for (int i = 0; i < 16; ++i) bm[i] = Ws[cb + ln][ks * 32 + g * 16 + i];
      c8 = __builtin_amdgcn_wmma_f32_16x16x32_f16(false, a, false, bm, (short)0, c8, false, false);
    }
    int oc = cb + ln;
    float s = 0.f, q = 0.f;
#pragma unroll
    for (int r = 0; r < 8; ++r) {
      float v = c8[r];
      s += v; q += v * v;
      Cs[g * 8 + r][oc] = (_Float16)v;
    }
    atomicAdd(&st[0][oc], s);
    atomicAdd(&st[1][oc], q);
    __syncthreads();
    // coalesced b128 store of the 16x128 tile
    {
      int m0 = (mtile0 + it) * 16;
      int r  = tid >> 4;
      int co = (tid & 15) * 8;
      *(uint4*)&Y[(size_t)(m0 + r) * Nout + obase + co] = *(const uint4*)&Cs[r][co];
    }
    wait_async0();
    __syncthreads();
  }
  if (tid < 128) {
    atomicAdd(&stats[obase + tid], st[0][tid]);
    atomicAdd(&stats[Nout + obase + tid], st[1][tid]);
  }
}

// ---------------- BN stat finalize: scale/shift per channel ----------------
__global__ void bn_finalize(const float* __restrict__ stats, const float* __restrict__ gamma,
                            const float* __restrict__ beta, float* __restrict__ s,
                            float* __restrict__ t, int Cn, float invM) {
  int c = blockIdx.x * blockDim.x + threadIdx.x;
  if (c < Cn) {
    float mean = stats[c] * invM;
    float var  = stats[Cn + c] * invM - mean * mean;
    float sc = gamma[c] * rsqrtf(var + EPSV);
    s[c] = sc; t[c] = beta[c] - mean * sc;
  }
}

// ---------------- in-place BN+lrelu on h1 (vectorized 8 halves/thread) ----------------
__global__ void act_kernel(_Float16* __restrict__ h, const float* __restrict__ s,
                           const float* __restrict__ t) {
  __shared__ float ss[C1], tt[C1];
  if (threadIdx.x < C1) { ss[threadIdx.x] = s[threadIdx.x]; tt[threadIdx.x] = t[threadIdx.x]; }
  __syncthreads();
  size_t i0 = ((size_t)blockIdx.x * 256 + threadIdx.x) * 8;
  uint4 v = *(const uint4*)(h + i0);
  _Float16* hp = (_Float16*)&v;
  int cb = (int)(i0 & (C1 - 1));
#pragma unroll
  for (int j = 0; j < 8; ++j) {
    float f = lrelu(ss[cb + j] * (float)hp[j] + tt[cb + j]);
    hp[j] = (_Float16)f;
  }
  *(uint4*)(h + i0) = v;
}

// ---------------- f32 -> f16 weight convert ----------------
__global__ void cvt_f16(const float* __restrict__ src, _Float16* __restrict__ dst, int n) {
  int i = blockIdx.x * 256 + threadIdx.x;
  if (i < n) dst[i] = (_Float16)src[i];
}

// ---------------- apply BN+lrelu then max over k ----------------
__global__ void bn_max_k(const _Float16* __restrict__ Y2, const float* __restrict__ s2,
                         const float* __restrict__ t2, _Float16* __restrict__ h2) {
  int tg = blockIdx.x * 256 + threadIdx.x;  // over M3*C2
  int c = tg % C2; int bn = tg / C2;
  float sc = s2[c], sh = t2[c];
  float m = -3.4e38f;
  size_t base = (size_t)bn * KNN * C2 + c;
  for (int k = 0; k < KNN; ++k) m = fmaxf(m, lrelu(sc * (float)Y2[base + (size_t)k * C2] + sh));
  h2[(size_t)bn * C2 + c] = (_Float16)m;
}

// ---------------- apply BN+lrelu then max over n ----------------
__global__ void bn_max_n(const _Float16* __restrict__ Y3, const float* __restrict__ s3,
                         const float* __restrict__ t3, float* __restrict__ gvec) {
  int tg = blockIdx.x * 256 + threadIdx.x;  // over B_*C3
  int c = tg % C3; int b = tg / C3;
  float sc = s3[c], sh = t3[c];
  float m = -3.4e38f;
  size_t base = (size_t)b * N_ * C3 + c;
  for (int n = 0; n < N_; ++n) m = fmaxf(m, lrelu(sc * (float)Y3[base + (size_t)n * C3] + sh));
  gvec[b * C3 + c] = m;
}

// ---------------- FC head (tiny, one block): fc1+bn4+fc2+bn5+fc3/fc4 ----------------
__global__ void head_kernel(const float* __restrict__ gvec,
                            const float* __restrict__ fc1_w, const float* __restrict__ bn4_g, const float* __restrict__ bn4_b,
                            const float* __restrict__ fc2_w, const float* __restrict__ bn5_g, const float* __restrict__ bn5_b,
                            const float* __restrict__ fc3_w, const float* __restrict__ fc3_b,
                            const float* __restrict__ fc4_w, const float* __restrict__ fc4_b,
                            float* __restrict__ matA, float* __restrict__ biasA) {
  __shared__ float G1[B_][1024];
  __shared__ float G2[B_][512];
  __shared__ float G3[B_][256];
  int tid = threadIdx.x;  // 512 threads
  for (int i = tid; i < B_ * 1024; i += 512) G1[i >> 10][i & 1023] = gvec[i];
  __syncthreads();
  {
    float acc[B_] = {};
    for (int c = 0; c < 1024; ++c) {
      float w = fc1_w[tid * 1024 + c];
#pragma unroll
      for (int b = 0; b < B_; ++b) acc[b] += G1[b][c] * w;
    }
    float mean = 0.f, var = 0.f;
#pragma unroll
    for (int b = 0; b < B_; ++b) mean += acc[b];
    mean *= (1.f / B_);
#pragma unroll
    for (int b = 0; b < B_; ++b) { float d = acc[b] - mean; var += d * d; }
    var *= (1.f / B_);
    float sc = bn4_g[tid] * rsqrtf(var + EPSV), sh = bn4_b[tid];
#pragma unroll
    for (int b = 0; b < B_; ++b) G2[b][tid] = lrelu(sc * (acc[b] - mean) + sh);
  }
  __syncthreads();
  if (tid < 256) {
    float acc[B_] = {};
    for (int c = 0; c < 512; ++c) {
      float w = fc2_w[tid * 512 + c];
#pragma unroll
      for (int b = 0; b < B_; ++b) acc[b] += G2[b][c] * w;
    }
    float mean = 0.f, var = 0.f;
#pragma unroll
    for (int b = 0; b < B_; ++b) mean += acc[b];
    mean *= (1.f / B_);
#pragma unroll
    for (int b = 0; b < B_; ++b) { float d = acc[b] - mean; var += d * d; }
    var *= (1.f / B_);
    float sc = bn5_g[tid] * rsqrtf(var + EPSV), sh = bn5_b[tid];
#pragma unroll
    for (int b = 0; b < B_; ++b) G3[b][tid] = lrelu(sc * (acc[b] - mean) + sh);
  }
  __syncthreads();
  if (tid < 72) {
    int b = tid / 9, e = tid % 9;
    float s = fc3_b[e];
    for (int c = 0; c < 256; ++c) s += fc3_w[e * 256 + c] * G3[b][c];
    matA[b * 9 + e] = s + ((e == 0 || e == 4 || e == 8) ? 1.f : 0.f);
  } else if (tid < 96) {
    int i = tid - 72, b = i / 3, e = i % 3;
    float s = fc4_b[e];
    for (int c = 0; c < 256; ++c) s += fc4_w[e * 256 + c] * G3[b][c];
    biasA[b * 3 + e] = s;
  }
}

// ---------------- apply learned 3x3 transform + bias ----------------
__global__ void transform_kernel(const float* __restrict__ x, const float* __restrict__ matA,
                                 const float* __restrict__ biasA, float* __restrict__ out) {
  int b = blockIdx.y;
  int n = blockIdx.x * 256 + threadIdx.x;
  float p0 = x[(b * C_ + 0) * N_ + n];
  float p1 = x[(b * C_ + 1) * N_ + n];
  float p2 = x[(b * C_ + 2) * N_ + n];
#pragma unroll
  for (int d = 0; d < 3; ++d) {
    float v = p0 * matA[b * 9 + 0 + d] + p1 * matA[b * 9 + 3 + d] + p2 * matA[b * 9 + 6 + d] + biasA[b * 3 + d];
    out[(size_t)(b * 3 + d) * N_ + n] = v;
  }
}

__global__ void zero_kernel(float* __restrict__ p, int n) {
  int i = blockIdx.x * 256 + threadIdx.x;
  if (i < n) p[i] = 0.f;
}

extern "C" void kernel_launch(void* const* d_in, const int* in_sizes, int n_in,
                              void* d_out, int out_size, void* d_ws, size_t ws_size,
                              hipStream_t stream) {
  (void)in_sizes; (void)n_in; (void)out_size; (void)ws_size;
  const float* x       = (const float*)d_in[0];
  const float* conv1_w = (const float*)d_in[1];
  const float* bn1_g   = (const float*)d_in[2];
  const float* bn1_b   = (const float*)d_in[3];
  const float* conv2_w = (const float*)d_in[4];
  const float* bn2_g   = (const float*)d_in[5];
  const float* bn2_b   = (const float*)d_in[6];
  const float* conv3_w = (const float*)d_in[7];
  const float* bn3_g   = (const float*)d_in[8];
  const float* bn3_b   = (const float*)d_in[9];
  const float* fc1_w   = (const float*)d_in[10];
  const float* bn4_g   = (const float*)d_in[11];
  const float* bn4_b   = (const float*)d_in[12];
  const float* fc2_w   = (const float*)d_in[13];
  const float* bn5_g   = (const float*)d_in[14];
  const float* bn5_b   = (const float*)d_in[15];
  const float* fc3_w   = (const float*)d_in[16];
  const float* fc3_b   = (const float*)d_in[17];
  const float* fc4_w   = (const float*)d_in[18];
  const float* fc4_b   = (const float*)d_in[19];
  float* out = (float*)d_out;

  // workspace carve-up (256B aligned)
  char* ws = (char*)d_ws;
  size_t off = 0;
  auto carve = [&](size_t bytes) { char* p = ws + off; off = (off + bytes + 255) & ~(size_t)255; return p; };
  int*       idx   = (int*)      carve((size_t)M1 * sizeof(int));
  _Float16*  h1    = (_Float16*) carve((size_t)M1 * C1 * sizeof(_Float16));
  _Float16*  y2    = (_Float16*) carve((size_t)M1 * C2 * sizeof(_Float16));
  _Float16*  h2    = (_Float16*) carve((size_t)M3 * C2 * sizeof(_Float16));
  _Float16*  y3    = (_Float16*) carve((size_t)M3 * C3 * sizeof(_Float16));
  _Float16*  w2h   = (_Float16*) carve((size_t)C2 * C1 * sizeof(_Float16));
  _Float16*  w3h   = (_Float16*) carve((size_t)C3 * C2 * sizeof(_Float16));
  float*     gvec  = (float*)    carve((size_t)B_ * C3 * sizeof(float));
  float*     stats = (float*)    carve((size_t)(2 * C1 + 2 * C2 + 2 * C3) * sizeof(float));
  float*     st1 = stats, *st2 = stats + 2 * C1, *st3 = stats + 2 * C1 + 2 * C2;
  float*     s1    = (float*)    carve((size_t)2 * (C1 + C2 + C3) * sizeof(float));
  float*     t1 = s1 + C1, *s2 = t1 + C1, *t2 = s2 + C2, *s3 = t2 + C2, *t3 = s3 + C3;
  float*     matA  = (float*)    carve((size_t)B_ * 9 * sizeof(float));
  float*     biasA = (float*)    carve((size_t)B_ * 3 * sizeof(float));

  int nstats = 2 * (C1 + C2 + C3);
  zero_kernel<<<(nstats + 255) / 256, 256, 0, stream>>>(stats, nstats);

  cvt_f16<<<(C2 * C1 + 255) / 256, 256, 0, stream>>>(conv2_w, w2h, C2 * C1);
  cvt_f16<<<(C3 * C2 + 255) / 256, 256, 0, stream>>>(conv3_w, w3h, C3 * C2);

  knn_kernel<<<dim3(N_ / 256, B_), 256, 0, stream>>>(x, idx);

  conv1_kernel<<<M1 / (16 * 8), 256, 0, stream>>>(x, idx, conv1_w, h1, st1);
  bn_finalize<<<1, 256, 0, stream>>>(st1, bn1_g, bn1_b, s1, t1, C1, 1.f / (float)M1);

  act_kernel<<<(int)((size_t)M1 * C1 / 8 / 256), 256, 0, stream>>>(h1, s1, t1);

  gemm_async<64, 8><<<dim3(M1 / (16 * 8), 1), 256, 0, stream>>>(h1, w2h, y2, st2, C2);
  bn_finalize<<<1, 256, 0, stream>>>(st2, bn2_g, bn2_b, s2, t2, C2, 1.f / (float)M1);

  bn_max_k<<<(M3 * C2) / 256, 256, 0, stream>>>(y2, s2, t2, h2);

  gemm_async<128, 8><<<dim3(M3 / (16 * 8), C3 / 128), 256, 0, stream>>>(h2, w3h, y3, st3, C3);
  bn_finalize<<<(C3 + 255) / 256, 256, 0, stream>>>(st3, bn3_g, bn3_b, s3, t3, C3, 1.f / (float)M3);

  bn_max_n<<<(B_ * C3) / 256, 256, 0, stream>>>(y3, s3, t3, gvec);

  head_kernel<<<1, 512, 0, stream>>>(gvec, fc1_w, bn4_g, bn4_b, fc2_w, bn5_g, bn5_b,
                                     fc3_w, fc3_b, fc4_w, fc4_b, matA, biasA);

  transform_kernel<<<dim3(N_ / 256, B_), 256, 0, stream>>>(x, matA, biasA, out);
}
